// MyNet_7026566496735
// MI455X (gfx1250) — compile-verified, matrix operand order checked by reading, not run on hardware
//
#include <hip/hip_runtime.h>
#include <hip/hip_bf16.h>
#include <math.h>

// ---------------------------------------------------------------------------
// Types for CDNA5 WMMA
// ---------------------------------------------------------------------------
typedef _Float16 v16h __attribute__((ext_vector_type(16)));
typedef _Float16 v8h  __attribute__((ext_vector_type(8)));
typedef float    v8f  __attribute__((ext_vector_type(8)));

#define N_NODES 100000
#define N_EDGES 800000
#define N_GRAPHS 64

// ---------------------------------------------------------------------------
// Generic fill
// ---------------------------------------------------------------------------
__global__ void fill_f32(float* __restrict__ p, float v, int n) {
    int i = blockIdx.x * blockDim.x + threadIdx.x;
    if (i < n) p[i] = v;
}

// ---------------------------------------------------------------------------
// f32 -> f16 with column padding (for GEMM A operand)
// ---------------------------------------------------------------------------
__global__ void cvt_pad_f16(const float* __restrict__ X, _Float16* __restrict__ A,
                            int M, int K, int Kpad) {
    int i = blockIdx.x * blockDim.x + threadIdx.x;
    if (i >= M * Kpad) return;
    int r = i / Kpad, c = i - r * Kpad;
    A[i] = (c < K) ? (_Float16)X[(size_t)r * K + c] : (_Float16)0.0f;
}

// ---------------------------------------------------------------------------
// Pack weight W[K,N] (row-major f32) into WMMA B-operand layout, f16.
// Layout: [kt][nt][lane(32)][16 halfs] ; lane&15 = column-in-tile,
// halfs h<8 -> Klocal = h + (lane>=16 ? 8:0), h>=8 -> Klocal = h+8 + (lane>=16?8:0)
// ---------------------------------------------------------------------------
__global__ void pack_b(const float* __restrict__ W, _Float16* __restrict__ Bp,
                       int K, int N, int Kt) {
    int NtTot = N >> 4;
    int total = Kt * NtTot * 32;
    int t = blockIdx.x * blockDim.x + threadIdx.x;
    if (t >= total) return;
    int lane = t & 31;
    int nt   = (t >> 5) % NtTot;
    int kt   = t / (32 * NtTot);
    int col  = nt * 16 + (lane & 15);
    int kadd = (lane >> 4) * 8;
    _Float16* out = Bp + (size_t)t * 16;
#pragma unroll
    for (int h = 0; h < 16; ++h) {
        int kl = (h < 8 ? h : h + 8) + kadd;
        int k  = kt * 32 + kl;
        float v = (k < K) ? W[(size_t)k * N + col] : 0.0f;
        out[h] = (_Float16)v;
    }
}

// ---------------------------------------------------------------------------
// WMMA GEMM: C[M,N] = A[M,Kpad] (f16, row-major) x Bpacked (f16)
// Block = 128 threads = 4 waves. The block cooperatively stages the B panel
// for its column group into LDS once (Kt*NT tiles); each wave then computes a
// 16 x (NT*16) strip for its own M-tile, reading B via ds_load_b128.
// Optional bias + ReLU epilogue, optional f32 and/or f16 outputs.
// ---------------------------------------------------------------------------
template <int NT>
__global__ void wmma_gemm(const _Float16* __restrict__ A,
                          const _Float16* __restrict__ Bp,
                          const float* __restrict__ bias,
                          float* __restrict__ Cf32,
                          _Float16* __restrict__ Cf16,
                          int Kt, int NtTot, int Mtiles, int relu) {
    extern __shared__ _Float16 Bs[];   // [Kt][NT][32][16]

    const int tid  = threadIdx.x;
    const int lane = tid & 31;
    const int wv   = tid >> 5;               // wave in block (0..3)
    const int ng   = blockIdx.y;             // column group

    // ---- cooperative B panel load: global -> LDS (16B chunks) ----
    {
        const int chunksPerTile = 512 / 8;   // 64 x uint4 per 32x16 tile-slice
        const int totalChunks   = Kt * NT * chunksPerTile;
        uint4* dstv = (uint4*)Bs;
        for (int t = tid; t < totalChunks; t += 128) {
            int kt = t / (NT * chunksPerTile);
            int r  = t - kt * NT * chunksPerTile;
            int j  = r / chunksPerTile;
            int c  = r - j * chunksPerTile;
            const uint4* srcv =
                (const uint4*)(Bp + ((size_t)kt * NtTot + (size_t)ng * NT + j) * 512);
            dstv[t] = srcv[c];
        }
    }
    __syncthreads();

    const int mtile = blockIdx.x * 4 + wv;   // wave-uniform
    if (mtile < Mtiles) {
        const int Kpad = Kt * 32;
        const int row  = mtile * 16 + (lane & 15);
        const int koff = (lane >> 4) * 8;

        v8f acc[NT];
#pragma unroll
        for (int j = 0; j < NT; ++j) acc[j] = (v8f)0.0f;

        const _Float16* arow = A + (size_t)row * Kpad;
        for (int kt = 0; kt < Kt; ++kt) {
            v8h a0 = *(const v8h*)(arow + kt * 32 + koff);
            v8h a1 = *(const v8h*)(arow + kt * 32 + koff + 16);
            v16h a = __builtin_shufflevector(a0, a1, 0, 1, 2, 3, 4, 5, 6, 7,
                                                     8, 9, 10, 11, 12, 13, 14, 15);
#pragma unroll
            for (int j = 0; j < NT; ++j) {
                v16h b = *(const v16h*)(Bs + (((size_t)kt * NT + j) * 32 + lane) * 16);
                acc[j] = __builtin_amdgcn_wmma_f32_16x16x32_f16(
                    false, a, false, b, (short)0, acc[j], false, false);
            }
        }

        const int mbase = mtile * 16 + (lane >> 4) * 8;
        const int col0  = lane & 15;
        const int N     = NtTot * 16;
#pragma unroll
        for (int j = 0; j < NT; ++j) {
            int col  = (ng * NT + j) * 16 + col0;
            float bv = bias ? bias[col] : 0.0f;
#pragma unroll
            for (int r = 0; r < 8; ++r) {
                float v = acc[j][r] + bv;
                if (relu) v = v > 0.0f ? v : 0.0f;
                size_t idx = (size_t)(mbase + r) * N + col;
                if (Cf32) Cf32[idx] = v;
                if (Cf16) Cf16[idx] = (_Float16)v;
            }
        }
    }
}

// ---------------------------------------------------------------------------
// Attention scores: s[n] = H[n,:].a_src , d[n] = H[n,:].a_dst  (wave per node)
// ---------------------------------------------------------------------------
__global__ void node_scores(const float* __restrict__ H,
                            const float* __restrict__ a_s,
                            const float* __restrict__ a_d,
                            float* __restrict__ s, float* __restrict__ d,
                            int N, int Do) {
    int wave = (blockIdx.x * blockDim.x + threadIdx.x) >> 5;
    int lane = threadIdx.x & 31;
    if (wave >= N) return;
    const float* h = H + (size_t)wave * Do;
    float ss = 0.0f, dd = 0.0f;
    for (int c = lane; c < Do; c += 32) {
        float hv = h[c];
        ss += hv * a_s[c];
        dd += hv * a_d[c];
    }
    for (int o = 16; o; o >>= 1) { ss += __shfl_xor(ss, o); dd += __shfl_xor(dd, o); }
    if (lane == 0) { s[wave] = ss; d[wave] = dd; }
}

// ---------------------------------------------------------------------------
// Float atomic max via sign-aware integer ordering (init must be -inf)
// ---------------------------------------------------------------------------
__device__ __forceinline__ void atomicMaxF32(float* addr, float val) {
    if (!signbit(val)) atomicMax((int*)addr, __float_as_int(val));
    else               atomicMin((unsigned int*)addr, __float_as_uint(val));
}

// Pass 1: e = leaky_relu(s[src]+d[dst]); segment max into m[dst]
__global__ void edge_pass1(const int* __restrict__ ei,
                           const float* __restrict__ s, const float* __restrict__ d,
                           float* __restrict__ eval, float* __restrict__ m,
                           int E, int Nn) {
    int e = blockIdx.x * blockDim.x + threadIdx.x;
    if (e >= E + Nn) return;
    int src = (e < E) ? ei[e]     : (e - E);
    int dst = (e < E) ? ei[E + e] : (e - E);
    float ev = s[src] + d[dst];
    ev = ev > 0.0f ? ev : 0.2f * ev;
    eval[e] = ev;
    atomicMaxF32(m + dst, ev);
}

// m = isfinite(m) ? m : 0   (only -inf possible)
__global__ void fix_m(float* __restrict__ m, int N) {
    int i = blockIdx.x * blockDim.x + threadIdx.x;
    if (i >= N) return;
    float v = m[i];
    m[i] = (v < -3.0e38f) ? 0.0f : v;
}

// Pass 2: ex = exp(e - m[dst]); den[dst] += ex ; store ex in place of e
__global__ void edge_pass2(const int* __restrict__ ei,
                           float* __restrict__ eval,
                           const float* __restrict__ m, float* __restrict__ den,
                           int E, int Nn) {
    int e = blockIdx.x * blockDim.x + threadIdx.x;
    if (e >= E + Nn) return;
    int dst = (e < E) ? ei[E + e] : (e - E);
    float ex = __expf(eval[e] - m[dst]);
    eval[e] = ex;
    atomicAdd(den + dst, ex);
}

// Pass 3 (wave per edge): Y[dst,:] += H[src,:] * (ex / (den[dst]+1e-16))
__global__ void edge_agg(const int* __restrict__ ei,
                         const float* __restrict__ eval, const float* __restrict__ den,
                         const float* __restrict__ H, float* __restrict__ Y,
                         int E, int Nn, int Do) {
    int wave = (blockIdx.x * blockDim.x + threadIdx.x) >> 5;
    int lane = threadIdx.x & 31;
    if (wave >= E + Nn) return;
    int src = (wave < E) ? ei[wave]     : (wave - E);
    int dst = (wave < E) ? ei[E + wave] : (wave - E);
    float alpha = eval[wave] / (den[dst] + 1e-16f);
    const float* h = H + (size_t)src * Do;
    float*       y = Y + (size_t)dst * Do;
    __builtin_prefetch(h, 0, 0);  // global_prefetch_b8 on gathered row
    for (int c = lane; c < Do; c += 32) atomicAdd(y + c, h[c] * alpha);
}

// Finalize layer: Y = relu(Y + b); also emit f16 copy for next GEMM
__global__ void finalize_layer(float* __restrict__ Y, const float* __restrict__ b,
                               _Float16* __restrict__ Y16, int M, int Do) {
    int i = blockIdx.x * blockDim.x + threadIdx.x;
    if (i >= M * Do) return;
    int c = i % Do;
    float v = Y[i] + b[c];
    v = v > 0.0f ? v : 0.0f;
    Y[i] = v;
    Y16[i] = (_Float16)v;
}

// global_add_pool
__global__ void pool_kernel(const float* __restrict__ out3, const int* __restrict__ batch,
                            float* __restrict__ pooled, int N, int Do) {
    int i = blockIdx.x * blockDim.x + threadIdx.x;
    if (i >= N * Do) return;
    int n = i / Do, c = i - n * Do;
    atomicAdd(pooled + (size_t)batch[n] * Do + c, out3[i]);
}

// Final policy layer: [N,32] x [32,1] + b  (wave per node)
__global__ void policy_dot(const _Float16* __restrict__ h, const float* __restrict__ w,
                           const float* __restrict__ b, float* __restrict__ out,
                           int N, int K) {
    int wave = (blockIdx.x * blockDim.x + threadIdx.x) >> 5;
    int lane = threadIdx.x & 31;
    if (wave >= N) return;
    float acc = 0.0f;
    for (int k = lane; k < K; k += 32) acc += (float)h[(size_t)wave * K + k] * w[k];
    for (int o = 16; o; o >>= 1) acc += __shfl_xor(acc, o);
    if (lane == 0) out[wave] = acc + b[0];
}

// Value MLP: 64 graphs, tiny — one block per graph
__global__ void value_mlp(const float* __restrict__ pooled,
                          const float* __restrict__ W0, const float* __restrict__ b0,
                          const float* __restrict__ W1, const float* __restrict__ b1,
                          const float* __restrict__ W2, const float* __restrict__ b2,
                          const float* __restrict__ W3, const float* __restrict__ b3,
                          float* __restrict__ out) {
    __shared__ float bufA[128];
    __shared__ float bufB[128];
    int g = blockIdx.x, t = threadIdx.x;
    bufA[t] = pooled[(size_t)g * 128 + t];
    __syncthreads();
    {   // 128 -> 128
        float acc = b0[t];
        for (int k = 0; k < 128; ++k) acc += bufA[k] * W0[(size_t)k * 128 + t];
        bufB[t] = acc > 0.0f ? acc : 0.0f;
    }
    __syncthreads();
    if (t < 64) {   // 128 -> 64
        float acc = b1[t];
        for (int k = 0; k < 128; ++k) acc += bufB[k] * W1[(size_t)k * 64 + t];
        bufA[t] = acc > 0.0f ? acc : 0.0f;
    }
    __syncthreads();
    if (t < 32) {   // 64 -> 32
        float acc = b2[t];
        for (int k = 0; k < 64; ++k) acc += bufA[k] * W2[(size_t)k * 32 + t];
        bufB[t] = acc > 0.0f ? acc : 0.0f;
    }
    __syncthreads();
    if (t == 0) {   // 32 -> 1
        float acc = b3[0];
        for (int k = 0; k < 32; ++k) acc += bufB[k] * W3[k];
        out[g] = acc;
    }
}

// ---------------------------------------------------------------------------
// Host launcher
// ---------------------------------------------------------------------------
static inline int cdiv(int a, int b) { return (a + b - 1) / b; }

extern "C" void kernel_launch(void* const* d_in, const int* in_sizes, int n_in,
                              void* d_out, int out_size, void* d_ws, size_t ws_size,
                              hipStream_t stream) {
    (void)in_sizes; (void)n_in; (void)out_size; (void)ws_size;
    const float* x   = (const float*)d_in[0];
    const int* ei    = (const int*)d_in[1];
    const int* batch = (const int*)d_in[2];
    const float* W1 = (const float*)d_in[3],  *as1 = (const float*)d_in[4],
               *ad1 = (const float*)d_in[5],  *b1  = (const float*)d_in[6];
    const float* W2 = (const float*)d_in[7],  *as2 = (const float*)d_in[8],
               *ad2 = (const float*)d_in[9],  *b2  = (const float*)d_in[10];
    const float* W3 = (const float*)d_in[11], *as3 = (const float*)d_in[12],
               *ad3 = (const float*)d_in[13], *b3  = (const float*)d_in[14];
    const float* pW0 = (const float*)d_in[15], *pb0 = (const float*)d_in[16];
    const float* pW1 = (const float*)d_in[17], *pb1 = (const float*)d_in[18];
    const float* pW2 = (const float*)d_in[19], *pb2 = (const float*)d_in[20];
    const float* pW3 = (const float*)d_in[21], *pb3 = (const float*)d_in[22];
    const float* vW0 = (const float*)d_in[23], *vb0 = (const float*)d_in[24];
    const float* vW1 = (const float*)d_in[25], *vb1 = (const float*)d_in[26];
    const float* vW2 = (const float*)d_in[27], *vb2 = (const float*)d_in[28];
    const float* vW3 = (const float*)d_in[29], *vb3 = (const float*)d_in[30];

    float* out_policy = (float*)d_out;                // [100000]
    float* out_value  = (float*)d_out + N_NODES;      // [64]

    // ---- workspace carve-up ----
    char* ws = (char*)d_ws;
    size_t off = 0;
    auto carve = [&](size_t bytes) { size_t o = off; off += (bytes + 255) & ~(size_t)255; return o; };
    float*    bufA  = (float*)   (ws + carve((size_t)N_NODES * 256 * 4));
    float*    bufB  = (float*)   (ws + carve((size_t)N_NODES * 256 * 4));
    _Float16* h16a  = (_Float16*)(ws + carve((size_t)N_NODES * 256 * 2));
    _Float16* h16b  = (_Float16*)(ws + carve((size_t)N_NODES * 256 * 2));
    float*    sbuf  = (float*)   (ws + carve((size_t)N_NODES * 4));
    float*    dbuf  = (float*)   (ws + carve((size_t)N_NODES * 4));
    float*    mbuf  = (float*)   (ws + carve((size_t)N_NODES * 4));
    float*    den   = (float*)   (ws + carve((size_t)N_NODES * 4));
    float*    eval  = (float*)   (ws + carve((size_t)(N_EDGES + N_NODES) * 4));
    float*    pooled= (float*)   (ws + carve((size_t)N_GRAPHS * 128 * 4));
    auto bp_bytes = [](int Kt, int N) { return (size_t)Kt * (N / 16) * 32 * 16 * 2; };
    _Float16* W1p  = (_Float16*)(ws + carve(bp_bytes(2, 256)));
    _Float16* W2p  = (_Float16*)(ws + carve(bp_bytes(8, 128)));
    _Float16* W3p  = (_Float16*)(ws + carve(bp_bytes(4, 128)));
    _Float16* pW0p = (_Float16*)(ws + carve(bp_bytes(4, 128)));
    _Float16* pW1p = (_Float16*)(ws + carve(bp_bytes(4, 64)));
    _Float16* pW2p = (_Float16*)(ws + carve(bp_bytes(2, 32)));

    const int Etot = N_EDGES + N_NODES;
    const int Mt   = N_NODES / 16;      // 6250

    // ---- pack all weights into WMMA B layout ----
    auto launch_pack = [&](const float* W, _Float16* Bp, int K, int N, int Kt) {
        int total = Kt * (N / 16) * 32;
        pack_b<<<cdiv(total, 256), 256, 0, stream>>>(W, Bp, K, N, Kt);
    };
    launch_pack(W1, W1p, 59, 256, 2);
    launch_pack(W2, W2p, 256, 128, 8);
    launch_pack(W3, W3p, 128, 128, 4);
    launch_pack(pW0, pW0p, 128, 128, 4);
    launch_pack(pW1, pW1p, 128, 64, 4);
    launch_pack(pW2, pW2p, 64, 32, 2);

    // ---- x -> f16 padded [N, 64] ----
    cvt_pad_f16<<<cdiv(N_NODES * 64, 256), 256, 0, stream>>>(x, h16a, N_NODES, 59, 64);

    // GEMM launcher: 4 waves/block, LDS = Kt*NT tiles of B (Kt*NT*1024 bytes)
    auto launch_gemm8 = [&](const _Float16* A16, const _Float16* Bp, const float* bias,
                            float* Cf32, _Float16* Cf16, int Kt, int NtTot, int relu) {
        dim3 grid(cdiv(Mt, 4), NtTot / 8);
        size_t lds = (size_t)Kt * 8 * 1024;
        wmma_gemm<8><<<grid, 128, lds, stream>>>(A16, Bp, bias, Cf32, Cf16,
                                                 Kt, NtTot, Mt, relu);
    };

    // ---- one GAT layer ----
    auto gat_layer = [&](const _Float16* A16, int Kt, const _Float16* Wp, int Do,
                         const float* a_s, const float* a_d, const float* bias,
                         float* H, float* Y, _Float16* Y16) {
        launch_gemm8(A16, Wp, nullptr, H, nullptr, Kt, Do / 16, 0);
        node_scores<<<cdiv(N_NODES * 32, 256), 256, 0, stream>>>(H, a_s, a_d, sbuf, dbuf, N_NODES, Do);
        fill_f32<<<cdiv(N_NODES, 256), 256, 0, stream>>>(mbuf, -INFINITY, N_NODES);
        fill_f32<<<cdiv(N_NODES, 256), 256, 0, stream>>>(den, 0.0f, N_NODES);
        fill_f32<<<cdiv(N_NODES * Do, 256), 256, 0, stream>>>(Y, 0.0f, N_NODES * Do);
        edge_pass1<<<cdiv(Etot, 256), 256, 0, stream>>>(ei, sbuf, dbuf, eval, mbuf, N_EDGES, N_NODES);
        fix_m<<<cdiv(N_NODES, 256), 256, 0, stream>>>(mbuf, N_NODES);
        edge_pass2<<<cdiv(Etot, 256), 256, 0, stream>>>(ei, eval, mbuf, den, N_EDGES, N_NODES);
        edge_agg<<<cdiv(Etot * 32, 256), 256, 0, stream>>>(ei, eval, den, H, Y, N_EDGES, N_NODES, Do);
        finalize_layer<<<cdiv(N_NODES * Do, 256), 256, 0, stream>>>(Y, bias, Y16, N_NODES, Do);
    };

    gat_layer(h16a, 2, W1p, 256, as1, ad1, b1, bufA, bufB, h16b);  // L1: 64->256
    gat_layer(h16b, 8, W2p, 128, as2, ad2, b2, bufA, bufB, h16a);  // L2: 256->128
    gat_layer(h16a, 4, W3p, 128, as3, ad3, b3, bufA, bufB, h16b);  // L3: 128->128
    // bufB = final node embedding f32 ; h16b = f16 copy

    // ---- global add pool ----
    fill_f32<<<cdiv(N_GRAPHS * 128, 256), 256, 0, stream>>>(pooled, 0.0f, N_GRAPHS * 128);
    pool_kernel<<<cdiv(N_NODES * 128, 256), 256, 0, stream>>>(bufB, batch, pooled, N_NODES, 128);

    // ---- policy MLP (WMMA, bias+ReLU fused, f16 chained) ----
    launch_gemm8(h16b, pW0p, pb0, nullptr, h16a, 4, 8, 1);                     // 128->128
    wmma_gemm<4><<<dim3(cdiv(Mt, 4), 1), 128, (size_t)4 * 4 * 1024, stream>>>(
        h16a, pW1p, pb1, nullptr, h16b, 4, 4, Mt, 1);                          // 128->64
    wmma_gemm<2><<<dim3(cdiv(Mt, 4), 1), 128, (size_t)2 * 2 * 1024, stream>>>(
        h16b, pW2p, pb2, nullptr, h16a, 2, 2, Mt, 1);                          // 64->32
    policy_dot<<<cdiv(N_NODES * 32, 256), 256, 0, stream>>>(h16a, pW3, pb3, out_policy, N_NODES, 32);

    // ---- value MLP ----
    value_mlp<<<N_GRAPHS, 128, 0, stream>>>(pooled, vW0, vb0, vW1, vb1, vW2, vb2, vW3, vb3, out_value);
}
